// QAGNN_Message_Passing_33088428048942
// MI455X (gfx1250) — matrix operand order repaired and checked by reading, not compiled, not run on hardware
//
#include <hip/hip_runtime.h>

// ---------------------------------------------------------------------------
// Types for CDNA5 WMMA (wave32): v_wmma_f32_16x16x32_f16
// ---------------------------------------------------------------------------
typedef _Float16 h8_t  __attribute__((ext_vector_type(8)));
typedef _Float16 h16_t __attribute__((ext_vector_type(16)));
typedef float    f8_t  __attribute__((ext_vector_type(8)));

union H16 { h16_t v; h8_t h[2]; };

// gfx1250 async global->LDS copy path (ASYNCcnt-tracked), if toolchain exposes it
#if defined(__has_builtin)
#if __has_builtin(__builtin_amdgcn_global_load_async_to_lds_b128)
#define USE_ASYNC_LDS 1
#endif
#endif
#ifndef USE_ASYNC_LDS
#define USE_ASYNC_LDS 0
#endif

#if USE_ASYNC_LDS
// signature (from toolchain diagnostic): (v4i AS1*, v4i AS3*, imm, imm)
typedef int v4i_b128 __attribute__((vector_size(4 * sizeof(int))));
typedef __attribute__((address_space(1))) v4i_b128* gptr_b128;
typedef __attribute__((address_space(3))) v4i_b128* lptr_b128;
#define GLD_ASYNC_B128(gsrc, ldst)                                            \
    __builtin_amdgcn_global_load_async_to_lds_b128(                           \
        (gptr_b128)(gsrc), (lptr_b128)(ldst), 0, 0)
__device__ __forceinline__ void wait_async_lds() {
#if __has_builtin(__builtin_amdgcn_s_wait_asynccnt)
    __builtin_amdgcn_s_wait_asynccnt(0);
#else
    asm volatile("s_wait_asynccnt 0x0" ::: "memory");
#endif
}
#endif

__device__ __forceinline__ float gelu_f(float x) {
    return 0.5f * x * (1.0f + erff(x * 0.7071067811865476f));
}

// monotonic float<->uint encoding for atomicMax-based segment max
__device__ __forceinline__ unsigned enc_f(float f) {
    int i = __float_as_int(f);
    return (i >= 0) ? ((unsigned)i ^ 0x80000000u) : ~(unsigned)i;
}
__device__ __forceinline__ float dec_f(unsigned u) {
    int i = (u & 0x80000000u) ? (int)(u ^ 0x80000000u) : (int)(~u);
    return __int_as_float(i);
}

// ---------------------------------------------------------------------------
// Generic f16 WMMA GEMM:  C = act(scale*(A @ B^T_stored + bias + addend))
//   A  : MxK row-major f16 (lda)
//   Bt : NcxK row-major f16 (weights pre-transposed; ldb)
//   C  : f32 out (optional), Ch : f16 out (optional)
// Block tile 128x128, BK=32, 256 threads = 8 waves, each wave 32x64.
// Double-buffered LDS, one barrier per K-iteration; tile k+1 streams into the
// alternate buffer (async-to-LDS when available) while tile k is computed.
// ---------------------------------------------------------------------------
#define BM 128
#define BN 128
#define BK 32

template<int ACT, bool HAS_BIAS, bool HAS_ADD>
__global__ __launch_bounds__(256) void gemm_kernel(
    const _Float16* __restrict__ A, int lda,
    const _Float16* __restrict__ Bt, int ldb,
    const float* __restrict__ bias,
    const float* __restrict__ addend, int ldadd,
    float scale,
    float* __restrict__ C, int ldc,
    _Float16* __restrict__ Ch, int ldch,
    int M, int K)
{
    __shared__ _Float16 sA[2][BM][BK + 8];   // +8 halves pad keeps 16B alignment
    __shared__ _Float16 sB[2][BN][BK + 8];

    const int tid  = threadIdx.x;
    const int m0   = blockIdx.y * BM;
    const int n0   = blockIdx.x * BN;
    const int w    = tid >> 5;
    const int lane = tid & 31;
    const int wm   = w >> 1;          // 0..3 -> 32-row strip
    const int wn   = w & 1;           // 0..1 -> 64-col strip
    const int mlo  = lane & 15;
    const int hs   = lane >> 4;       // half-wave select

    // per-thread staging slot: 16 halves of A and of B per tile
    const int row = tid >> 1;
    const int cs  = (tid & 1) * 16;
    const bool arow_ok = (m0 + row) < M;
    const _Float16* Abase = A + (size_t)(m0 + row) * lda + cs;
    const _Float16* Bbase = Bt + (size_t)(n0 + row) * ldb + cs;
    const int nk = K / BK;

    f8_t acc[2][4];
#pragma unroll
    for (int mt = 0; mt < 2; ++mt)
#pragma unroll
        for (int nt = 0; nt < 4; ++nt) { f8_t z = {}; acc[mt][nt] = z; }

#if USE_ASYNC_LDS
    // ---- prologue: stream tile 0 into buffer 0 ----
    if (arow_ok) {
        GLD_ASYNC_B128(Abase,     &sA[0][row][cs]);
        GLD_ASYNC_B128(Abase + 8, &sA[0][row][cs + 8]);
    } else {
        // zero OOB rows so LDS holds defined data (results guarded at store)
        h8_t z = {};
        *(h8_t*)&sA[0][row][cs]     = z;
        *(h8_t*)&sA[0][row][cs + 8] = z;
    }
    GLD_ASYNC_B128(Bbase,     &sB[0][row][cs]);
    GLD_ASYNC_B128(Bbase + 8, &sB[0][row][cs + 8]);
    wait_async_lds();
    __syncthreads();
#else
    {   // ---- prologue: stage tile 0 through registers ----
        h8_t a0 = {}, a1 = {};
        if (arow_ok) { a0 = *(const h8_t*)Abase; a1 = *(const h8_t*)(Abase + 8); }
        h8_t b0 = *(const h8_t*)Bbase, b1 = *(const h8_t*)(Bbase + 8);
        *(h8_t*)&sA[0][row][cs]     = a0;
        *(h8_t*)&sA[0][row][cs + 8] = a1;
        *(h8_t*)&sB[0][row][cs]     = b0;
        *(h8_t*)&sB[0][row][cs + 8] = b1;
        __syncthreads();
    }
#endif

    for (int it = 0; it < nk; ++it) {
        const int cur = it & 1;
        const int nxt = cur ^ 1;
        const bool have_next = (it + 1) < nk;

#if USE_ASYNC_LDS
        // kick off next tile into the alternate buffer (background copy)
        if (have_next) {
            const _Float16* ga = Abase + (size_t)(it + 1) * BK;
            const _Float16* gb = Bbase + (size_t)(it + 1) * BK;
            if (arow_ok) {
                GLD_ASYNC_B128(ga,     &sA[nxt][row][cs]);
                GLD_ASYNC_B128(ga + 8, &sA[nxt][row][cs + 8]);
            }
            GLD_ASYNC_B128(gb,     &sB[nxt][row][cs]);
            GLD_ASYNC_B128(gb + 8, &sB[nxt][row][cs + 8]);
        }
#else
        h8_t na0 = {}, na1 = {}, nb0 = {}, nb1 = {};
        if (have_next) {
            const _Float16* ga = Abase + (size_t)(it + 1) * BK;
            const _Float16* gb = Bbase + (size_t)(it + 1) * BK;
            if (arow_ok) { na0 = *(const h8_t*)ga; na1 = *(const h8_t*)(ga + 8); }
            nb0 = *(const h8_t*)gb; nb1 = *(const h8_t*)(gb + 8);
        }
#endif

        // ---- compute on current buffer ----
        H16 a[2], b[4];
#pragma unroll
        for (int mt = 0; mt < 2; ++mt) {
            const int r = wm * 32 + mt * 16 + mlo;
            a[mt].h[0] = *(const h8_t*)&sA[cur][r][hs * 8];
            a[mt].h[1] = *(const h8_t*)&sA[cur][r][16 + hs * 8];
        }
#pragma unroll
        for (int nt = 0; nt < 4; ++nt) {
            const int r = wn * 64 + nt * 16 + mlo;
            b[nt].h[0] = *(const h8_t*)&sB[cur][r][hs * 8];
            b[nt].h[1] = *(const h8_t*)&sB[cur][r][16 + hs * 8];
        }
#pragma unroll
        for (int mt = 0; mt < 2; ++mt)
#pragma unroll
            for (int nt = 0; nt < 4; ++nt)
                acc[mt][nt] = __builtin_amdgcn_wmma_f32_16x16x32_f16(
                    false, a[mt].v, false, b[nt].v,
                    (short)0, acc[mt][nt], false, false);

        if (have_next) {
#if USE_ASYNC_LDS
            wait_async_lds();        // next-tile copies landed in LDS
#else
            *(h8_t*)&sA[nxt][row][cs]     = na0;
            *(h8_t*)&sA[nxt][row][cs + 8] = na1;
            *(h8_t*)&sB[nxt][row][cs]     = nb0;
            *(h8_t*)&sB[nxt][row][cs + 8] = nb1;
#endif
            __syncthreads();
        }
    }

    // ---- epilogue ----
#pragma unroll
    for (int mt = 0; mt < 2; ++mt) {
#pragma unroll
        for (int nt = 0; nt < 4; ++nt) {
            const int col = n0 + wn * 64 + nt * 16 + mlo;
            const float bv = HAS_BIAS ? bias[col] : 0.0f;
#pragma unroll
            for (int v = 0; v < 8; ++v) {
                const int r = m0 + wm * 32 + mt * 16 + hs * 8 + v;
                if (r < M) {
                    float val = acc[mt][nt][v] + bv;
                    if (HAS_ADD) val += addend[(size_t)r * ldadd + col];
                    val *= scale;
                    if (ACT == 1) val = fmaxf(val, 0.0f);
                    if (ACT == 2) val = gelu_f(val);
                    if (C)  C[(size_t)r * ldc + col]   = val;
                    if (Ch) Ch[(size_t)r * ldch + col] = (_Float16)val;
                }
            }
        }
    }
}

// ---------------------------------------------------------------------------
// Elementwise / gather / scatter kernels
// ---------------------------------------------------------------------------
__global__ void cvt_h_kernel(const float* __restrict__ in, _Float16* __restrict__ out, size_t count) {
    size_t i = (size_t)blockIdx.x * blockDim.x + threadIdx.x;
    if (i < count) out[i] = (_Float16)in[i];
}

__global__ void transpose_f16_kernel(const float* __restrict__ W, _Float16* __restrict__ Wt,
                                     int K, int Nc) {
    int idx = blockIdx.x * 256 + threadIdx.x;
    if (idx < K * Nc) {
        int k = idx / Nc, n = idx % Nc;
        Wt[(size_t)n * K + k] = (_Float16)W[idx];
    }
}

// nfe[:, :128] = gelu(W_ntype[nt] + b_ntype);  Bsin = f16(sin(1.1^j * score))
__global__ void node_prep_kernel(const int* __restrict__ node_type,
                                 const float* __restrict__ node_score,
                                 const float* __restrict__ W_ntype,
                                 const float* __restrict__ b_ntype,
                                 float* __restrict__ nfe,
                                 _Float16* __restrict__ Bsin) {
    const int n = blockIdx.x;      // node
    const int j = threadIdx.x;     // 0..127
    const int nt = node_type[n];
    nfe[(size_t)n * 256 + j] = gelu_f(W_ntype[nt * 128 + j] + b_ntype[j]);
    const float s = node_score[n];
    Bsin[(size_t)n * 128 + j] = (_Float16)sinf(powf(1.1f, (float)j) * s);
}

// X2 = f16(concat(X, nfe)) per node (512 cols)
__global__ void build_x2_kernel(const float* __restrict__ X, const float* __restrict__ nfe,
                                _Float16* __restrict__ X2, size_t count) {
    size_t i = (size_t)blockIdx.x * blockDim.x + threadIdx.x;
    if (i >= count) return;
    size_t node = i >> 9;          // /512
    int c = (int)(i & 511);
    float v = (c < 256) ? X[node * 256 + c] : nfe[node * 256 + (c - 256)];
    X2[i] = (_Float16)v;
}

// hedge = f16(relu(We1[et] + We1[39+nt_src] + We1[43+nt_dst] + be1)) ; multi-hot GEMM as 3 row gathers
__global__ void edge_feat_kernel(const int* __restrict__ edge_index,
                                 const int* __restrict__ edge_type,
                                 const int* __restrict__ node_type,
                                 const float* __restrict__ We1,
                                 const float* __restrict__ be1,
                                 _Float16* __restrict__ hedge, int E_) {
    const int e = blockIdx.x;
    const int d = threadIdx.x;     // 0..255
    int r0, r1, r2;
    if (e < E_) {
        int s = edge_index[e], t = edge_index[E_ + e];
        r0 = edge_type[e];
        r1 = 39 + node_type[s];
        r2 = 43 + node_type[t];
    } else {
        int nt = node_type[e - E_];
        r0 = 38; r1 = 39 + nt; r2 = 43 + nt;
    }
    float v = We1[r0 * 256 + d] + We1[r1 * 256 + d] + We1[r2 * 256 + d] + be1[d];
    hedge[(size_t)e * 256 + d] = (_Float16)fmaxf(v, 0.0f);
}

__global__ void deg_kernel(const int* __restrict__ edge_index, float* __restrict__ deg,
                           int E_, int EN_) {
    int e = blockIdx.x * 256 + threadIdx.x;
    if (e >= EN_) return;
    int s = (e < E_) ? edge_index[e] : (e - E_);
    atomicAdd(&deg[s], 1.0f);
}

// pass1: per-edge per-head score = sum(q_src * (k_dst + k_edge)), + segment max by src
__global__ void scores_kernel(const int* __restrict__ edge_index,
                              const float* __restrict__ XQ, const float* __restrict__ XK,
                              const float* __restrict__ EK,
                              float* __restrict__ scores, unsigned* __restrict__ smax,
                              int E_, int EN_) {
    const int wid = threadIdx.x >> 5, lane = threadIdx.x & 31;
    const int e = blockIdx.x * 8 + wid;
    if (e >= EN_) return;
    int s, d;
    if (e < E_) { s = edge_index[e]; d = edge_index[E_ + e]; } else { s = d = e - E_; }
    const float* q  = XQ + (size_t)s * 256;
    const float* k1 = XK + (size_t)d * 256;
    const float* k2 = EK + (size_t)e * 256;
    const int base = lane * 8;
    float acc = 0.0f;
#pragma unroll
    for (int i = 0; i < 8; ++i) acc += q[base + i] * (k1[base + i] + k2[base + i]);
    acc += __shfl_xor(acc, 1, 32);
    acc += __shfl_xor(acc, 2, 32);
    acc += __shfl_xor(acc, 4, 32);
    if ((lane & 7) == 0) {
        const int h = lane >> 3;
        scores[(size_t)e * 4 + h] = acc;
        atomicMax(&smax[(size_t)s * 4 + h], enc_f(acc));
    }
}

// pass2: ex = exp(score - max), segment sum by src (ex overwrites scores)
__global__ void exp_sum_kernel(const int* __restrict__ edge_index,
                               float* __restrict__ scores,
                               const unsigned* __restrict__ smax,
                               float* __restrict__ ssum, int E_, int EN_) {
    int idx = blockIdx.x * 256 + threadIdx.x;
    if (idx >= EN_ * 4) return;
    const int e = idx >> 2, h = idx & 3;
    const int s = (e < E_) ? edge_index[e] : (e - E_);
    const float m = dec_f(smax[s * 4 + h]);
    const float ex = expf(scores[idx] - m);
    scores[idx] = ex;
    atomicAdd(&ssum[s * 4 + h], ex);
}

// pass3: aggr[dst] += (XM[src] + EM[e]) * alpha;  alpha = ex/(sum+eps) * deg[src]
__global__ void aggr_kernel(const int* __restrict__ edge_index,
                            const float* __restrict__ ex, const float* __restrict__ ssum,
                            const float* __restrict__ deg,
                            const float* __restrict__ XM, const float* __restrict__ EM,
                            float* __restrict__ aggr, int E_, int EN_) {
    const int wid = threadIdx.x >> 5, lane = threadIdx.x & 31;
    const int e = blockIdx.x * 8 + wid;
    if (e >= EN_) return;
    int s, d;
    if (e < E_) { s = edge_index[e]; d = edge_index[E_ + e]; } else { s = d = e - E_; }
    const int h = lane >> 3;
    const float alpha = ex[(size_t)e * 4 + h] / (ssum[(size_t)s * 4 + h] + 1e-16f) * deg[s];
    const int base = lane * 8;
    const float* xm = XM + (size_t)s * 256;
    const float* em = EM + (size_t)e * 256;
    float* ag = aggr + (size_t)d * 256;
#pragma unroll
    for (int i = 0; i < 8; ++i)
        atomicAdd(&ag[base + i], (xm[base + i] + em[base + i]) * alpha);
}

// ---------------------------------------------------------------------------
// Host-side dispatch
// ---------------------------------------------------------------------------
static void launch_gemm(hipStream_t st,
                        const _Float16* A, int lda, const _Float16* Bt, int ldb,
                        const float* bias, const float* addend, int ldadd, float scale,
                        float* C, int ldc, _Float16* Ch, int ldch,
                        int M, int Nc, int K, int act) {
    dim3 grid(Nc / 128, (M + 127) / 128);
    dim3 block(256);
    const bool hb = bias != nullptr, ha = addend != nullptr;
#define GL(a, b, c) gemm_kernel<a, b, c><<<grid, block, 0, st>>>( \
        A, lda, Bt, ldb, bias, addend, ldadd, scale, C, ldc, Ch, ldch, M, K)
    if (act == 0 && hb && !ha)       GL(0, true, false);
    else if (act == 0 && !hb && !ha) GL(0, false, false);
    else if (act == 1 && hb && !ha)  GL(1, true, false);
    else if (act == 2 && hb && !ha)  GL(2, true, false);
    else if (act == 2 && hb && ha)   GL(2, true, true);
#undef GL
}

extern "C" void kernel_launch(void* const* d_in, const int* in_sizes, int n_in,
                              void* d_out, int out_size, void* d_ws, size_t ws_size,
                              hipStream_t stream) {
    (void)in_sizes; (void)n_in; (void)out_size; (void)ws_size;
    constexpr int N = 25600, E = 100000, EN = 125600;

    const float* H          = (const float*)d_in[0];
    const float* node_score = (const float*)d_in[1];
    const int*   edge_index = (const int*)d_in[2];
    const int*   edge_type  = (const int*)d_in[3];
    const int*   node_type  = (const int*)d_in[4];
    const float* W_ntype = (const float*)d_in[5];
    const float* b_ntype = (const float*)d_in[6];
    const float* W_score = (const float*)d_in[7];
    const float* b_score = (const float*)d_in[8];
    const float* We1 = (const float*)d_in[9];
    const float* be1 = (const float*)d_in[10];
    const float* We2 = (const float*)d_in[11];
    const float* be2 = (const float*)d_in[12];
    const float* Wk  = (const float*)d_in[13];
    const float* bk  = (const float*)d_in[14];
    const float* Wm  = (const float*)d_in[15];
    const float* bm  = (const float*)d_in[16];
    const float* Wq  = (const float*)d_in[17];
    const float* bq  = (const float*)d_in[18];
    const float* Wmlp1 = (const float*)d_in[19];
    const float* bmlp1 = (const float*)d_in[20];
    const float* Wmlp2 = (const float*)d_in[21];
    const float* bmlp2 = (const float*)d_in[22];
    const float* Wvh = (const float*)d_in[23];
    const float* bvh = (const float*)d_in[24];
    const float* Wvx = (const float*)d_in[25];
    const float* bvx = (const float*)d_in[26];
    float* OUT = (float*)d_out;

    // ---- workspace carve-up ----
    char* base = (char*)d_ws;
    size_t off = 0;
    auto alloc = [&](size_t bytes) -> void* {
        void* r = base + off;
        off += (bytes + 255) & ~(size_t)255;
        return r;
    };
    _Float16* Hh    = (_Float16*)alloc((size_t)N * 256 * 2);
    _Float16* Bsin  = (_Float16*)alloc((size_t)N * 128 * 2);
    float*    nfe   = (float*)   alloc((size_t)N * 256 * 4);
    float*    X     = (float*)   alloc((size_t)N * 256 * 4);
    _Float16* Xh    = (_Float16*)alloc((size_t)N * 256 * 2);
    _Float16* X2    = (_Float16*)alloc((size_t)N * 512 * 2);
    _Float16* hedge = (_Float16*)alloc((size_t)EN * 256 * 2);
    _Float16* eembh = (_Float16*)alloc((size_t)EN * 256 * 2);
    float*    XK    = (float*)   alloc((size_t)N * 256 * 4);
    float*    XM    = (float*)   alloc((size_t)N * 256 * 4);
    float*    XQ    = (float*)   alloc((size_t)N * 256 * 4);
    float*    EK    = (float*)   alloc((size_t)EN * 256 * 4);
    float*    EM    = (float*)   alloc((size_t)EN * 256 * 4);
    float*    scores= (float*)   alloc((size_t)EN * 4 * 4);
    unsigned* smax  = (unsigned*)alloc((size_t)N * 4 * 4);
    float*    ssum  = (float*)   alloc((size_t)N * 4 * 4);
    float*    deg   = (float*)   alloc((size_t)N * 4);
    float*    aggr  = (float*)   alloc((size_t)N * 256 * 4);
    _Float16* aggrh = (_Float16*)alloc((size_t)N * 256 * 2);
    _Float16* h1h   = (_Float16*)alloc((size_t)N * 256 * 2);
    float*    tmp   = (float*)   alloc((size_t)N * 256 * 4);
    _Float16* Wscore_t = (_Float16*)alloc(128 * 128 * 2);
    _Float16* We2_t    = (_Float16*)alloc(256 * 256 * 2);
    _Float16* Wvh_t    = (_Float16*)alloc(256 * 256 * 2);
    _Float16* Wvx_t    = (_Float16*)alloc(256 * 256 * 2);
    _Float16* WkA_t[2], *WkB_t[2], *WmA_t[2], *WmB_t[2], *Wq_t[2], *W1_t[2], *W2_t[2];
    for (int i = 0; i < 2; ++i) {
        WkA_t[i] = (_Float16*)alloc(256 * 512 * 2);
        WkB_t[i] = (_Float16*)alloc(256 * 256 * 2);
        WmA_t[i] = (_Float16*)alloc(256 * 512 * 2);
        WmB_t[i] = (_Float16*)alloc(256 * 256 * 2);
        Wq_t[i]  = (_Float16*)alloc(256 * 512 * 2);
        W1_t[i]  = (_Float16*)alloc(256 * 256 * 2);
        W2_t[i]  = (_Float16*)alloc(256 * 256 * 2);
    }

    auto tr = [&](const float* W, _Float16* Wt, int K, int Nc) {
        int cnt = K * Nc;
        transpose_f16_kernel<<<(cnt + 255) / 256, 256, 0, stream>>>(W, Wt, K, Nc);
    };

    // ---- weight prep (f16, transposed [N][K]) ----
    tr(W_score, Wscore_t, 128, 128);
    tr(We2, We2_t, 256, 256);
    tr(Wvh, Wvh_t, 256, 256);
    tr(Wvx, Wvx_t, 256, 256);
    for (int i = 0; i < 2; ++i) {
        tr(Wk + (size_t)i * 768 * 256,             WkA_t[i], 512, 256);
        tr(Wk + (size_t)i * 768 * 256 + 512 * 256, WkB_t[i], 256, 256);
        tr(Wm + (size_t)i * 768 * 256,             WmA_t[i], 512, 256);
        tr(Wm + (size_t)i * 768 * 256 + 512 * 256, WmB_t[i], 256, 256);
        tr(Wq + (size_t)i * 512 * 256,             Wq_t[i],  512, 256);
        tr(Wmlp1 + (size_t)i * 256 * 256,          W1_t[i],  256, 256);
        tr(Wmlp2 + (size_t)i * 256 * 256,          W2_t[i],  256, 256);
    }

    // ---- node / edge feature embeddings ----
    cvt_h_kernel<<<((size_t)N * 256 + 255) / 256, 256, 0, stream>>>(H, Hh, (size_t)N * 256);
    node_prep_kernel<<<N, 128, 0, stream>>>(node_type, node_score, W_ntype, b_ntype, nfe, Bsin);
    // nfe[:, 128:256] = gelu(Bsin @ W_score + b_score)
    launch_gemm(stream, Bsin, 128, Wscore_t, 128, b_score, nullptr, 0, 1.0f,
                nfe + 128, 256, nullptr, 0, N, 128, 128, /*gelu*/2);
    // edge MLP layer 1 as gathers, layer 2 as GEMM -> f16 edge_emb
    edge_feat_kernel<<<EN, 256, 0, stream>>>(edge_index, edge_type, node_type, We1, be1, hedge, E);
    launch_gemm(stream, hedge, 256, We2_t, 256, be2, nullptr, 0, 1.0f,
                nullptr, 0, eembh, 256, EN, 256, 256, /*none*/0);
    // degree by source (incl. self-loops)
    (void)hipMemsetAsync(deg, 0, (size_t)N * 4, stream);
    deg_kernel<<<(EN + 255) / 256, 256, 0, stream>>>(edge_index, deg, E, EN);

    const float* Xin = H;   // layer-0 input
    for (int i = 0; i < 2; ++i) {
        build_x2_kernel<<<((size_t)N * 512 + 255) / 256, 256, 0, stream>>>(
            Xin, nfe, X2, (size_t)N * 512);
        // node-side projections (bias folded here)
        launch_gemm(stream, X2, 512, WkA_t[i], 512, bk + i * 256, nullptr, 0, 1.0f,
                    XK, 256, nullptr, 0, N, 256, 512, 0);
        launch_gemm(stream, X2, 512, WmA_t[i], 512, bm + i * 256, nullptr, 0, 1.0f,
                    XM, 256, nullptr, 0, N, 256, 512, 0);
        launch_gemm(stream, X2, 512, Wq_t[i], 512, bq + i * 256, nullptr, 0, 0.125f,
                    XQ, 256, nullptr, 0, N, 256, 512, 0);      // * 1/sqrt(DPH)
        // edge-side projections
        launch_gemm(stream, eembh, 256, WkB_t[i], 256, nullptr, nullptr, 0, 1.0f,
                    EK, 256, nullptr, 0, EN, 256, 256, 0);
        launch_gemm(stream, eembh, 256, WmB_t[i], 256, nullptr, nullptr, 0, 1.0f,
                    EM, 256, nullptr, 0, EN, 256, 256, 0);
        // attention: scores -> segment softmax by src -> weighted scatter-add to dst
        (void)hipMemsetAsync(smax, 0, (size_t)N * 4 * 4, stream);
        (void)hipMemsetAsync(ssum, 0, (size_t)N * 4 * 4, stream);
        scores_kernel<<<(EN + 7) / 8, 256, 0, stream>>>(
            edge_index, XQ, XK, EK, scores, smax, E, EN);
        exp_sum_kernel<<<(EN * 4 + 255) / 256, 256, 0, stream>>>(
            edge_index, scores, smax, ssum, E, EN);
        (void)hipMemsetAsync(aggr, 0, (size_t)N * 256 * 4, stream);
        aggr_kernel<<<(EN + 7) / 8, 256, 0, stream>>>(
            edge_index, scores, ssum, deg, XM, EM, aggr, E, EN);
        // node MLP: X = gelu(relu(aggr@W1+b1)@W2+b2)
        cvt_h_kernel<<<((size_t)N * 256 + 255) / 256, 256, 0, stream>>>(
            aggr, aggrh, (size_t)N * 256);
        launch_gemm(stream, aggrh, 256, W1_t[i], 256, bmlp1 + i * 256, nullptr, 0, 1.0f,
                    nullptr, 0, h1h, 256, N, 256, 256, /*relu*/1);
        launch_gemm(stream, h1h, 256, W2_t[i], 256, bmlp2 + i * 256, nullptr, 0, 1.0f,
                    X, 256, Xh, 256, N, 256, 256, /*gelu*/2);
        Xin = X;
    }

    // out = gelu(H @ Wvh + bvh + X @ Wvx + bvx)
    launch_gemm(stream, Hh, 256, Wvh_t, 256, bvh, nullptr, 0, 1.0f,
                tmp, 256, nullptr, 0, N, 256, 256, 0);
    launch_gemm(stream, Xh, 256, Wvx_t, 256, bvx, tmp, 256, 1.0f,
                OUT, 256, nullptr, 0, N, 256, 256, /*gelu*/2);
}
